// ShiftedWindowAttention_80900003988203
// MI455X (gfx1250) — matrix operand order repaired and checked by reading, not compiled, not run on hardware
//
#include <hip/hip_runtime.h>

typedef __attribute__((ext_vector_type(16))) _Float16     v16h;
typedef __attribute__((ext_vector_type(8)))  float        v8f;
typedef __attribute__((ext_vector_type(4)))  unsigned int u32x4;

#define DIM    384
#define HEADS  12
#define HD     32
#define NTOK   49
#define NP     64
#define WSZ    7
#define SHIFT  3
#define IMG    56
#define BATCH  32
#define TOT_WIN (BATCH * 64)
#define SCALE  0.17677669529663687f   // 1/sqrt(32)

// LDS strides (in halfs) — chosen so 16-lane b128 accesses are bank-conflict free
#define XS_S 392
#define QK_S 40
#define VT_S 72
#define PB_S 72
#define OB_S 392
#define WB_S 40

#define XS_H (NTOK * XS_S)      // 19208 halfs : x tile 49x384 (f16)
#define QH_H (NP * QK_S)        //  2560 halfs : q or k per head, 64x32
#define VT_H (HD * VT_S)        //  2304 halfs : v transposed, 32x64
#define PB_H (16 * PB_S)        //  1152 halfs : P strip per wave, 16x64
#define OB_H (NTOK * OB_S)      // 19208 halfs : attention output 49x384
#define WB_SLAB (192 * WB_S)    //  7680 halfs : one async-staged weight K-slab
#define WB_H (2 * WB_SLAB)      // 15360 halfs : double buffer (proj: 384 rows)
#define BT_H (169 * HEADS)      //  2028 halfs : relative-position bias table

#define SM_HALFS (XS_H + 4*QH_H + 2*VT_H + 8*PB_H + OB_H + WB_H + BT_H)
#define SMEM_BYTES (SM_HALFS * 2)   // 159,736 B -> 2 blocks / 320KB WGP

union FragU { v16h h; u32x4 q[2]; };

// A-matrix fragment (16x32 f16): lane = row (lane&15), K split by lane-half:
// VGPR0..3 hold K = half*8 .. +7 ; VGPR4..7 hold K = 16+half*8 .. +7
__device__ __forceinline__ v16h ldA(const _Float16* base, int rs, int lane) {
  int hf = (lane >> 4) & 1, lq = lane & 15;
  const _Float16* p = base + lq * rs + hf * 8;
  FragU f;
  f.q[0] = *(const u32x4*)(p);
  f.q[1] = *(const u32x4*)(p + 16);
  return f.h;
}
// B-matrix fragment (32x16 f16): lane = column (lane&15), K = half*8 .. +7, 16+half*8 .. +7
// buffer layout: [output-column][K] (column contiguous in K)
__device__ __forceinline__ v16h ldB(const _Float16* base, int rs, int lane) {
  int hf = (lane >> 4) & 1, lq = lane & 15;
  const _Float16* p = base + lq * rs + hf * 8;
  FragU f;
  f.q[0] = *(const u32x4*)(p);
  f.q[1] = *(const u32x4*)(p + 16);
  return f.h;
}
__device__ __forceinline__ v8f wmma_f16(v16h a, v16h b, v8f c) {
  return __builtin_amdgcn_wmma_f32_16x16x32_f16(false, a, false, b,
                                                (short)0, c, false, false);
}

// ---- CDNA5 async global->LDS DMA (ASYNCcnt) -------------------------------
__device__ __forceinline__ void async_ld_b128(_Float16* lds_dst,
                                              const _Float16* gsrc) {
  unsigned l = (unsigned)(unsigned long long)lds_dst;  // low 32b = LDS offset
  asm volatile("global_load_async_to_lds_b128 %0, %1, off"
               :: "v"(l), "v"(gsrc) : "memory");
}
__device__ __forceinline__ void wait_async0() {
  asm volatile("s_wait_asynccnt 0x0" ::: "memory");
}
__device__ __forceinline__ void wait_async3() {   // newest 3 may be in flight
  asm volatile("s_wait_asynccnt 0x3" ::: "memory");
}

// stage one 32-wide K-slab of the 2-head qkv weight slice: 192 rows x 64B
__device__ __forceinline__ void stage_qkv_slab(_Float16* wb,
                                               const _Float16* WQ,
                                               int h0, int kc, int tid) {
#pragma unroll
  for (int t = 0; t < 3; ++t) {
    int i = tid + t * 256;          // 768 x 16B chunks
    int row = i >> 2, c = i & 3;    // row 0..191, 16B chunk 0..3
    int g = row >> 5, rin = row & 31;       // group: q0,q1,k0,k1,v0,v1
    int part = g >> 1, hh = g & 1;
    int grow = part * DIM + (h0 + hh) * HD + rin;
    async_ld_b128(wb + row * WB_S + c * 8,
                  WQ + grow * DIM + kc * 32 + c * 8);
  }
}
// stage one 32-wide K-slab of proj weights: 384 rows x 64B (whole WB region)
__device__ __forceinline__ void stage_proj_slab(_Float16* wb,
                                                const _Float16* WP,
                                                int kc, int tid) {
#pragma unroll
  for (int t = 0; t < 6; ++t) {
    int i = tid + t * 256;          // 1536 x 16B chunks
    int row = i >> 2, c = i & 3;
    async_ld_b128(wb + row * WB_S + c * 8,
                  WP + row * DIM + kc * 32 + c * 8);
  }
}

// ---------------------------------------------------------------------------
// One-time f32 -> f16 weight conversion into workspace (L2-resident, 1.2 MB)
// ---------------------------------------------------------------------------
__global__ void convert_w_kernel(const float* __restrict__ qkv_w,
                                 const float* __restrict__ proj_w,
                                 _Float16* __restrict__ wq,
                                 _Float16* __restrict__ wp) {
  int i = blockIdx.x * 256 + threadIdx.x;
  const int n1 = 3 * DIM * DIM;
  const int n2 = DIM * DIM;
  if (i < n1)            wq[i]      = (_Float16)qkv_w[i];
  else if (i < n1 + n2)  wp[i - n1] = (_Float16)proj_w[i - n1];
}

// ---------------------------------------------------------------------------
// Fully fused per-window shifted attention. Block = 256 thr (8 wave32),
// grid = 2048 windows. Weights DMA-staged to LDS per K-slab (ASYNCcnt).
// ---------------------------------------------------------------------------
__global__ __launch_bounds__(256)
void swin_attn_kernel(const float*    __restrict__ x,
                      const _Float16* __restrict__ WQ,     // [1152][384] f16
                      const _Float16* __restrict__ WP,     // [384][384]  f16
                      const float*    __restrict__ qkvB,   // [1152]
                      const float*    __restrict__ projB,  // [384]
                      const float*    __restrict__ btab,   // [169][12]
                      float*          __restrict__ out) {
  extern __shared__ __align__(16) char smem[];
  _Float16* XS  = (_Float16*)smem;         // [49][XS_S]
  _Float16* QH  = XS + XS_H;               // [2][64][QK_S] (scaled q)
  _Float16* KH  = QH + 2 * QH_H;           // [2][64][QK_S]
  _Float16* VT  = KH + 2 * QH_H;           // [2][32][VT_S] (v transposed)
  _Float16* PB  = VT + 2 * VT_H;           // [8][16][PB_S] (P strips)
  _Float16* OB  = PB + 8 * PB_H;           // [49][OB_S]    (attn output)
  _Float16* WB  = OB + OB_H;               // [2][192][WB_S] staged weights
  _Float16* BTH = WB + WB_H;               // bias table (f16)

  const int win = blockIdx.x;
  const int b   = win >> 6;
  const int wy  = (win >> 3) & 7;
  const int wx  = win & 7;
  const int tid = threadIdx.x;
  const int w   = tid >> 5;
  const int lane = tid & 31;
  const int hf  = (lane >> 4) & 1;
  const int lq  = lane & 15;
  const int mt  = w >> 1;            // row-tile for GEMM phases
  const int sel = w & 1;             // column interleave
  const bool arow_ok = (mt * 16 + lq) < NTOK;   // A-fragment row guard

  // ---- Phase 1: gather shifted window into LDS (f16) ----------------------
  for (int i = tid; i < NTOK * DIM; i += 256) {
    int n = i / DIM, c = i - n * DIM;
    int nh = n / WSZ, nw = n - nh * WSZ;
    int hs = wy * WSZ + nh + SHIFT; if (hs >= IMG) hs -= IMG;  // roll(-3)
    int ws = wx * WSZ + nw + SHIFT; if (ws >= IMG) ws -= IMG;
    XS[n * XS_S + c] = (_Float16)x[((b * IMG + hs) * IMG + ws) * DIM + c];
  }
  for (int i = tid; i < BT_H; i += 256) BTH[i] = (_Float16)btab[i];
  __syncthreads();

  // ---- Phase 2: per head-pair ---------------------------------------------
  for (int hp = 0; hp < HEADS / 2; ++hp) {
    const int h0 = hp * 2;

    // -- 2a: QKV GEMM, weights async-staged + double buffered ---------------
    {
      v8f acc[6] = {};
      stage_qkv_slab(WB, WQ, h0, 0, tid);            // prologue -> buf0
      for (int kc = 0; kc < DIM / 32; ++kc) {
        _Float16* cur = WB + (kc & 1) * WB_SLAB;
        if (kc + 1 < DIM / 32) {
          stage_qkv_slab(WB + ((kc + 1) & 1) * WB_SLAB, WQ, h0, kc + 1, tid);
          wait_async3();                             // current slab landed
        } else {
          wait_async0();
        }
        __syncthreads();                             // slab visible block-wide
        v16h a = {};
        if (arow_ok) a = ldA(XS + mt * 16 * XS_S + kc * 32, XS_S, lane);
#pragma unroll
        for (int j = 0; j < 6; ++j) {
          int nt = sel + 2 * j;                      // n-tile 0..11
          int g  = (nt >> 2) * 2 + ((nt >> 1) & 1);  // staged group
          int sub = (nt & 1) * 16;
          v16h bb = ldB(cur + (g * 32 + sub) * WB_S, WB_S, lane);
          acc[j] = wmma_f16(a, bb, acc[j]);
        }
        __syncthreads();                             // protect buffer reuse
      }
#pragma unroll
      for (int j = 0; j < 6; ++j) {
        int nt = sel + 2 * j;
        int part = nt >> 2;
        int hh   = (nt >> 1) & 1;
        int sub  = (nt & 1) * 16;
        int wrow = part * DIM + (h0 + hh) * HD + sub;
        float bv = qkvB[wrow + lq];
        int lc = sub + lq;
#pragma unroll
        for (int r = 0; r < 8; ++r) {
          int row = mt * 16 + r + hf * 8;
          float v = acc[j][r] + bv;
          if (part == 0)      QH[hh * QH_H + row * QK_S + lc] = (_Float16)(v * SCALE);
          else if (part == 1) KH[hh * QH_H + row * QK_S + lc] = (_Float16)v;
          else                VT[hh * VT_H + lc * VT_S + row] = (_Float16)v;
        }
      }
    }
    __syncthreads();

    // -- 2b: attention; wave -> (head hh, 16-row strip s) -------------------
    {
      const int hh = w >> 2;
      const int s  = w & 3;
      const int m0 = s * 16;
      const int head = h0 + hh;

      v16h aq = ldA(QH + hh * QH_H + m0 * QK_S, QK_S, lane);
      v8f sacc[4] = {};
#pragma unroll
      for (int ct = 0; ct < 4; ++ct) {
        v16h bk = ldB(KH + hh * QH_H + ct * 16 * QK_S, QK_S, lane);
        sacc[ct] = wmma_f16(aq, bk, sacc[ct]);
      }
      // relative-position bias + shift mask + pad-column mask
#pragma unroll
      for (int ct = 0; ct < 4; ++ct) {
        int mc = ct * 16 + lq;                       // key token
        if (mc >= NTOK) {
#pragma unroll
          for (int r = 0; r < 8; ++r) sacc[ct][r] = -1e30f;
        } else {
          int mh = mc / WSZ, mw = mc - mh * WSZ;
          int tmh = wy * WSZ + mh, tmw = wx * WSZ + mw;
          int regm = (tmh < 49 ? 0 : (tmh < 53 ? 1 : 2)) * 3
                   + (tmw < 49 ? 0 : (tmw < 53 ? 1 : 2));
#pragma unroll
          for (int r = 0; r < 8; ++r) {
            int nr = m0 + r + hf * 8;                // query token
            if (nr < NTOK) {
              int nh = nr / WSZ, nw = nr - nh * WSZ;
              int idx = (nh - mh + 6) * 13 + (nw - mw + 6);
              float bv = (float)BTH[idx * HEADS + head];
              int tnh = wy * WSZ + nh, tnw = wx * WSZ + nw;
              int regn = (tnh < 49 ? 0 : (tnh < 53 ? 1 : 2)) * 3
                       + (tnw < 49 ? 0 : (tnw < 53 ? 1 : 2));
              sacc[ct][r] += bv + ((regm != regn) ? -100.f : 0.f);
            }
          }
        }
      }
      // softmax: each row lives in one 16-lane half -> xor-shuffle reduce
#pragma unroll
      for (int r = 0; r < 8; ++r) {
        float mx = fmaxf(fmaxf(sacc[0][r], sacc[1][r]),
                         fmaxf(sacc[2][r], sacc[3][r]));
        for (int d = 1; d < 16; d <<= 1) mx = fmaxf(mx, __shfl_xor(mx, d, 32));
        float sum = 0.f;
#pragma unroll
        for (int ct = 0; ct < 4; ++ct) {
          float e = __expf(sacc[ct][r] - mx);
          sacc[ct][r] = e; sum += e;
        }
        for (int d = 1; d < 16; d <<= 1) sum += __shfl_xor(sum, d, 32);
        float inv = 1.0f / sum;
#pragma unroll
        for (int ct = 0; ct < 4; ++ct) sacc[ct][r] *= inv;
      }
      // stash P strip (wave-private LDS) in A-friendly row-major layout
      _Float16* pw = PB + w * PB_H;
#pragma unroll
      for (int ct = 0; ct < 4; ++ct)
#pragma unroll
        for (int r = 0; r < 8; ++r)
          pw[(r + hf * 8) * PB_S + ct * 16 + lq] = (_Float16)sacc[ct][r];

      // O strip = P(16x64) @ V(64x32); B from transposed v (contiguous in K)
      v8f oacc[2] = {};
#pragma unroll
      for (int kc = 0; kc < 2; ++kc) {
        v16h ap = ldA(pw + kc * 32, PB_S, lane);
#pragma unroll
        for (int nt = 0; nt < 2; ++nt) {
          v16h bv = ldB(VT + hh * VT_H + nt * 16 * VT_S + kc * 32, VT_S, lane);
          oacc[nt] = wmma_f16(ap, bv, oacc[nt]);
        }
      }
#pragma unroll
      for (int nt = 0; nt < 2; ++nt)
#pragma unroll
        for (int r = 0; r < 8; ++r) {
          int row = m0 + r + hf * 8;
          if (row < NTOK)
            OB[row * OB_S + head * HD + nt * 16 + lq] = (_Float16)oacc[nt][r];
        }
    }
    __syncthreads();
  }

  // ---- Phase 3: projection (49x384)@(384x384)^T, staged weights ----------
  {
    v8f pacc[12] = {};
    for (int kc = 0; kc < DIM / 32; ++kc) {
      stage_proj_slab(WB, WP, kc, tid);
      wait_async0();
      __syncthreads();
      v16h a = {};
      if (arow_ok) a = ldA(OB + mt * 16 * OB_S + kc * 32, OB_S, lane);
#pragma unroll
      for (int j = 0; j < 12; ++j) {
        int nt = sel + 2 * j;
        v16h bb = ldB(WB + nt * 16 * WB_S, WB_S, lane);
        pacc[j] = wmma_f16(a, bb, pacc[j]);
      }
      __syncthreads();
    }
#pragma unroll
    for (int j = 0; j < 12; ++j) {
      int ncol = (sel + 2 * j) * 16 + lq;
      float pb = projB[ncol];
#pragma unroll
      for (int r = 0; r < 8; ++r) {
        int row = mt * 16 + r + hf * 8;
        if (row < NTOK) {
          int nh = row / WSZ, nw = row - nh * WSZ;
          int ho = wy * WSZ + nh + SHIFT; if (ho >= IMG) ho -= IMG;  // roll(+3)
          int wo = wx * WSZ + nw + SHIFT; if (wo >= IMG) wo -= IMG;
          out[((b * IMG + ho) * IMG + wo) * DIM + ncol] = pacc[j][r] + pb;
        }
      }
    }
  }
}

// ---------------------------------------------------------------------------
extern "C" void kernel_launch(void* const* d_in, const int* in_sizes, int n_in,
                              void* d_out, int out_size, void* d_ws, size_t ws_size,
                              hipStream_t stream) {
  const float* x      = (const float*)d_in[0];
  const float* qkv_w  = (const float*)d_in[1];
  const float* qkv_b  = (const float*)d_in[2];
  const float* proj_w = (const float*)d_in[3];
  const float* proj_b = (const float*)d_in[4];
  const float* btab   = (const float*)d_in[5];

  _Float16* wq16 = (_Float16*)d_ws;                 // 1152*384 f16
  _Float16* wp16 = wq16 + 3 * DIM * DIM;            //  384*384 f16

  int ntot = 3 * DIM * DIM + DIM * DIM;
  convert_w_kernel<<<(ntot + 255) / 256, 256, 0, stream>>>(qkv_w, proj_w,
                                                           wq16, wp16);
  swin_attn_kernel<<<TOT_WIN, 256, SMEM_BYTES, stream>>>(x, wq16, wp16,
                                                         qkv_b, proj_b, btab,
                                                         (float*)d_out);
}
